// ReaderNet_12695923327082
// MI455X (gfx1250) — matrix operand order repaired and checked by reading, not compiled, not run on hardware
//
#include <hip/hip_runtime.h>
#include <hip/hip_bf16.h>
#include <math.h>

// ---------------------------------------------------------------------------
// Problem constants (match the reference)
// ---------------------------------------------------------------------------
static constexpr int kV    = 50257;
static constexpr int kVpad = 50432;          // = 197 * 256, multiple of 256
static constexpr int kH    = 256;
static constexpr int kB    = 128;
static constexpr int kS    = 50;
static constexpr int kT    = 32;
static constexpr int kTQ   = 32;
static constexpr int kH3   = 3 * kH;         // 768

typedef __attribute__((ext_vector_type(16))) __bf16 v16bf;
typedef __attribute__((ext_vector_type(8)))  float  v8f;

__device__ __forceinline__ __bf16 f2bf(float f) {
    unsigned u = __float_as_uint(f);
    unsigned r = (u + 0x7FFFu + ((u >> 16) & 1u)) >> 16;   // round-to-nearest-even
    unsigned short s = (unsigned short)r;
    return __builtin_bit_cast(__bf16, s);
}

__device__ __forceinline__ float sigmoidf_(float x) {
    return 1.0f / (1.0f + __expf(-x));
}

// Load A fragment (16x32 bf16, row-major src) for lane: row mr, K base k0.
__device__ __forceinline__ v16bf load_a_frag(const __bf16* __restrict__ A,
                                             int mr, int K, int k0, int half) {
    v16bf a;
    const __bf16* Ap = A + (size_t)mr * K + k0 + 16 * half;
#pragma unroll
    for (int i = 0; i < 8; ++i)  a[i] = Ap[i];          // K = k0+16*half+ 0..7
#pragma unroll
    for (int i = 8; i < 16; ++i) a[i] = Ap[i];          // K = k0+16*half+ 8..15
    return a;
}

// Load B fragment (32x16 bf16) from W[N,K] row-major: output col nr, K base k0.
__device__ __forceinline__ v16bf load_b_frag(const __bf16* __restrict__ W,
                                             int nr, int K, int k0, int half) {
    v16bf b;
    const __bf16* Wp = W + (size_t)nr * K + k0 + 16 * half;
#pragma unroll
    for (int i = 0; i < 16; ++i) b[i] = Wp[i];          // K = k0+16*half+ 0..15
    return b;
}

// ---------------------------------------------------------------------------
// f32 -> bf16 conversion with zero padding (elements [n, npad) -> 0)
// ---------------------------------------------------------------------------
__global__ void f2bf_pad_kernel(const float* __restrict__ src, __bf16* __restrict__ dst,
                                int n, int npad) {
    int i = blockIdx.x * blockDim.x + threadIdx.x;
    if (i < npad) dst[i] = (i < n) ? f2bf(src[i]) : f2bf(0.0f);
}

// ---------------------------------------------------------------------------
// Context embedding + position encoding: ctx[b,s,h] = sum_t emb[id,h]*l[t,h]
// grid = B*S blocks, 256 threads (h)
// ---------------------------------------------------------------------------
__global__ void embed_pos_kernel(const int* __restrict__ contexts,
                                 const float* __restrict__ emb,
                                 __bf16* __restrict__ ctx_bf) {
    const int bs = blockIdx.x;                 // b*S + s
    const int h  = threadIdx.x;                // 0..255
    const int* tok = contexts + (size_t)bs * kT;
    const float e = (float)h * (1.0f / (float)(kH - 1));
    float acc = 0.0f;
#pragma unroll
    for (int t = 0; t < kT; ++t) {
        const float sf = (float)t * (1.0f / (float)(kT - 1));
        const float l  = 1.0f - sf - e * (1.0f - 2.0f * sf);
        acc += emb[(size_t)tok[t] * kH + h] * l;
    }
    ctx_bf[(size_t)bs * kH + h] = f2bf(acc);
}

// Question embedding gather: grid = B*TQ, 256 threads
__global__ void embed_q_kernel(const int* __restrict__ questions,
                               const float* __restrict__ emb,
                               __bf16* __restrict__ qemb_bf) {
    const int row = blockIdx.x;
    const int h   = threadIdx.x;
    qemb_bf[(size_t)row * kH + h] = f2bf(emb[(size_t)questions[row] * kH + h]);
}

// ---------------------------------------------------------------------------
// WMMA GEMM: C[M,N] = A[M,K](bf16 row-major) * W[Npad,K]^T + bias[N]
// All fragment loads are unconditionally valid (W padded to >= grid.x*256 rows,
// M a multiple of 32). Only the final store is N-guarded.
// block = 128 threads (4 waves). Each wave owns a 32x64 output block:
// 2 M-tiles x 4 N-tiles = 8 accumulators; per K-step 2 A-frags + 4 B-frags
// feed 8 v_wmma ops. grid.x = ceil(N/256), grid.y = M/32.
// ---------------------------------------------------------------------------
__global__ void wmma_gemm_bias_kernel(const __bf16* __restrict__ A,
                                      const __bf16* __restrict__ W,
                                      const float* __restrict__ bias,
                                      float* __restrict__ C,
                                      int M, int N, int K) {
    const int lane = threadIdx.x & 31;
    const int wv   = threadIdx.x >> 5;
    const int half = lane >> 4;
    const int l16  = lane & 15;
    const int n0   = blockIdx.x * 256 + wv * 64;   // wave's first output column
    const int m0   = blockIdx.y * 32;              // wave's first output row

    v8f acc[2][4] = {};
    for (int k0 = 0; k0 < K; k0 += 32) {
        v16bf a0 = load_a_frag(A, m0      + l16, K, k0, half);
        v16bf a1 = load_a_frag(A, m0 + 16 + l16, K, k0, half);
#pragma unroll
        for (int nt = 0; nt < 4; ++nt) {
            v16bf b = load_b_frag(W, n0 + nt * 16 + l16, K, k0, half);
            acc[0][nt] = __builtin_amdgcn_wmma_f32_16x16x32_bf16(
                false, a0, false, b, (short)0, acc[0][nt], false, false);
            acc[1][nt] = __builtin_amdgcn_wmma_f32_16x16x32_bf16(
                false, a1, false, b, (short)0, acc[1][nt], false, false);
        }
    }

#pragma unroll
    for (int nt = 0; nt < 4; ++nt) {
        const int ncol = n0 + nt * 16 + l16;
        if (ncol < N) {
            const float bv = bias ? bias[ncol] : 0.0f;
#pragma unroll
            for (int mt = 0; mt < 2; ++mt) {
#pragma unroll
                for (int i = 0; i < 8; ++i) {
                    const int m = m0 + mt * 16 + i + 8 * half;
                    C[(size_t)m * N + ncol] = acc[mt][nt][i] + bv;
                }
            }
        }
    }
}

// ---------------------------------------------------------------------------
// One GRU timestep (recurrent part), fused: gh = h @ Whh^T (WMMA, LDS-staged),
// then gate math, h update, bf16 re-encode, optional sequence-output store.
// grid = B/16 = 8 blocks, block = 512 threads (16 waves, 3 N-tiles each).
// A-fragment (hidden state) is loaded once per K-step and reused for 3 WMMAs.
// ---------------------------------------------------------------------------
__global__ void gru_step_kernel(float* h_f32,
                                __bf16* h_bf,
                                const __bf16* __restrict__ Whh,
                                const float* __restrict__ bhh,
                                const float* __restrict__ gi_t,   // (b=0, step, col 0)
                                long gi_bstride,                  // floats between batches
                                float* out_t,                     // may be null
                                long out_bstride) {
    __shared__ float gh[16 * kH3];             // 48 KB
    const int lane = threadIdx.x & 31;
    const int wv   = threadIdx.x >> 5;         // 0..15
    const int half = lane >> 4;
    const int l16  = lane & 15;
    const int m0   = blockIdx.x * 16;

    // Prefetch the gi rows needed by the gate phase while the GEMM runs
    // (global_prefetch_b8; 16 rows x 768 f32, one prefetch per 32 floats).
    {
        const int pid = threadIdx.x;           // 0..511; need 16*24 = 384 prefetches
        if (pid < 16 * 24) {
            const int pr = pid / 24;           // local row
            const int pc = (pid % 24) * 32;    // column (floats)
            __builtin_prefetch(gi_t + (size_t)(m0 + pr) * gi_bstride + pc, 0, 1);
        }
    }

    // ---- GEMM phase: 16 waves x 3 tiles = 48 N-tiles covering 768 columns
    v8f acc[3] = {};
    const int mr = m0 + l16;
    for (int k0 = 0; k0 < kH; k0 += 32) {
        v16bf a = load_a_frag(h_bf, mr, kH, k0, half);
#pragma unroll
        for (int tt = 0; tt < 3; ++tt) {
            const int nr = (wv * 3 + tt) * 16 + l16;
            v16bf b = load_b_frag(Whh, nr, kH, k0, half);
            acc[tt] = __builtin_amdgcn_wmma_f32_16x16x32_bf16(
                false, a, false, b, (short)0, acc[tt], false, false);
        }
    }
#pragma unroll
    for (int tt = 0; tt < 3; ++tt) {
        const int ncol = (wv * 3 + tt) * 16 + l16;
#pragma unroll
        for (int i = 0; i < 8; ++i)
            gh[(i + 8 * half) * kH3 + ncol] = acc[tt][i];
    }
    __syncthreads();

    // ---- Gate phase: 16 rows x 256 cols = 4096 elems over 512 threads
    for (int idx = threadIdx.x; idx < 16 * kH; idx += blockDim.x) {
        const int r  = idx >> 8;               // local row
        const int hc = idx & 255;
        const int m  = m0 + r;                 // batch index
        const float* gip = gi_t + (size_t)m * gi_bstride;
        const float hr = gh[r * kH3 + hc]          + bhh[hc];
        const float hz = gh[r * kH3 + kH  + hc]    + bhh[kH  + hc];
        const float hn = gh[r * kH3 + 2*kH + hc]   + bhh[2*kH + hc];
        const float ir = gip[hc];
        const float iz = gip[kH  + hc];
        const float in = gip[2*kH + hc];
        const float rg = sigmoidf_(ir + hr);
        const float zg = sigmoidf_(iz + hz);
        const float ng = tanhf(in + rg * hn);
        const float hp = h_f32[(size_t)m * kH + hc];
        const float hv = (1.0f - zg) * ng + zg * hp;
        h_f32[(size_t)m * kH + hc] = hv;
        h_bf [(size_t)m * kH + hc] = f2bf(hv);
        if (out_t) out_t[(size_t)m * out_bstride + hc] = hv;
    }
}

// ---------------------------------------------------------------------------
// Zero-init hidden states (f32 + bf16)
// ---------------------------------------------------------------------------
__global__ void init_h_kernel(float* hf, __bf16* hb, int n) {
    int i = blockIdx.x * blockDim.x + threadIdx.x;
    if (i < n) { hf[i] = 0.0f; hb[i] = f2bf(0.0f); }
}

// facts = out_f + out_b
__global__ void add_kernel(const float* __restrict__ a, const float* __restrict__ b,
                           float* __restrict__ c, int n) {
    int i = blockIdx.x * blockDim.x + threadIdx.x;
    if (i < n) c[i] = a[i] + b[i];
}

// ---------------------------------------------------------------------------
// Attention + memory read + z = tanh(q * read) in bf16. grid = B, 256 threads.
// ---------------------------------------------------------------------------
__global__ void attn_kernel(const float* __restrict__ facts,
                            const float* __restrict__ q,
                            __bf16* __restrict__ z_bf) {
    const int b   = blockIdx.x;
    const int tid = threadIdx.x;
    __shared__ float qsh[kH];
    __shared__ float attn[kS];
    __shared__ float ssum;
    qsh[tid] = q[(size_t)b * kH + tid];
    __syncthreads();
    if (tid < kS) {
        const float* fr = facts + ((size_t)b * kS + tid) * kH;
        float e = 0.0f;
        for (int h = 0; h < kH; ++h) e += fr[h] * qsh[h];
        attn[tid] = e;
    }
    __syncthreads();
    if (tid == 0) {
        float m = attn[0];
        for (int s = 1; s < kS; ++s) m = fmaxf(m, attn[s]);
        float sum = 0.0f;
        for (int s = 0; s < kS; ++s) { float v = __expf(attn[s] - m); attn[s] = v; sum += v; }
        ssum = sum;
    }
    __syncthreads();
    const float inv = 1.0f / ssum;
    const float* fb = facts + (size_t)b * kS * kH;
    float acc = 0.0f;
    for (int s = 0; s < kS; ++s) acc += attn[s] * fb[(size_t)s * kH + tid];
    acc *= inv;
    z_bf[(size_t)b * kH + tid] = f2bf(tanhf(qsh[tid] * acc));
}

// ---------------------------------------------------------------------------
// Host-side orchestration
// ---------------------------------------------------------------------------
extern "C" void kernel_launch(void* const* d_in, const int* in_sizes, int n_in,
                              void* d_out, int out_size, void* d_ws, size_t ws_size,
                              hipStream_t stream) {
    const int*   contexts  = (const int*)  d_in[0];
    const int*   questions = (const int*)  d_in[1];
    const float* emb       = (const float*)d_in[2];
    const float* Wih_f = (const float*)d_in[3];
    const float* Whh_f = (const float*)d_in[4];
    const float* bih_f = (const float*)d_in[5];
    const float* bhh_f = (const float*)d_in[6];
    const float* Wih_b = (const float*)d_in[7];
    const float* Whh_b = (const float*)d_in[8];
    const float* bih_b = (const float*)d_in[9];
    const float* bhh_b = (const float*)d_in[10];
    const float* Wih_q = (const float*)d_in[11];
    const float* Whh_q = (const float*)d_in[12];
    const float* bih_q = (const float*)d_in[13];
    const float* bhh_q = (const float*)d_in[14];
    const float* Wo    = (const float*)d_in[15];
    const float* bo    = (const float*)d_in[16];

    // ---- workspace carve-up
    char* ws = (char*)d_ws;
    size_t off = 0;
    auto carve = [&](size_t bytes) -> char* {
        char* p = ws + off;
        off += (bytes + 255) & ~(size_t)255;
        return p;
    };
    const size_t nWg = (size_t)kH3 * kH;               // 768*256
    __bf16* ctx_bf  = (__bf16*)carve((size_t)kB * kS * kH * 2);
    __bf16* qemb_bf = (__bf16*)carve((size_t)kB * kTQ * kH * 2);
    __bf16* wihf_bf = (__bf16*)carve(nWg * 2);
    __bf16* whhf_bf = (__bf16*)carve(nWg * 2);
    __bf16* wihb_bf = (__bf16*)carve(nWg * 2);
    __bf16* whhb_bf = (__bf16*)carve(nWg * 2);
    __bf16* wihq_bf = (__bf16*)carve(nWg * 2);
    __bf16* whhq_bf = (__bf16*)carve(nWg * 2);
    __bf16* wo_bf   = (__bf16*)carve((size_t)kVpad * kH * 2);   // padded rows zeroed
    float*  gi_f    = (float*)carve((size_t)kB * kS * kH3 * 4);
    float*  gi_b    = (float*)carve((size_t)kB * kS * kH3 * 4);
    float*  gi_q    = (float*)carve((size_t)kB * kTQ * kH3 * 4);
    float*  hf_all  = (float*)carve((size_t)3 * kB * kH * 4);
    __bf16* hb_all  = (__bf16*)carve((size_t)3 * kB * kH * 2);
    float*  out_f   = (float*)carve((size_t)kB * kS * kH * 4);
    float*  out_b   = (float*)carve((size_t)kB * kS * kH * 4);
    float*  facts   = (float*)carve((size_t)kB * kS * kH * 4);
    __bf16* z_bf    = (__bf16*)carve((size_t)kB * kH * 2);
    (void)ws_size; (void)n_in; (void)in_sizes; (void)out_size;

    float*  hf_f = hf_all;            __bf16* hb_f = hb_all;
    float*  hf_b = hf_all + kB * kH;  __bf16* hb_b = hb_all + kB * kH;
    float*  hf_q = hf_all + 2*kB*kH;  __bf16* hb_q = hb_all + 2*kB*kH;

    // ---- 1. weight conversion to bf16 (Wo padded with zero rows)
    const int nW = (int)nWg;
    f2bf_pad_kernel<<<(nW + 255) / 256, 256, 0, stream>>>(Wih_f, wihf_bf, nW, nW);
    f2bf_pad_kernel<<<(nW + 255) / 256, 256, 0, stream>>>(Whh_f, whhf_bf, nW, nW);
    f2bf_pad_kernel<<<(nW + 255) / 256, 256, 0, stream>>>(Wih_b, wihb_bf, nW, nW);
    f2bf_pad_kernel<<<(nW + 255) / 256, 256, 0, stream>>>(Whh_b, whhb_bf, nW, nW);
    f2bf_pad_kernel<<<(nW + 255) / 256, 256, 0, stream>>>(Wih_q, wihq_bf, nW, nW);
    f2bf_pad_kernel<<<(nW + 255) / 256, 256, 0, stream>>>(Whh_q, whhq_bf, nW, nW);
    const int nWo    = kV * kH;
    const int nWoPad = kVpad * kH;
    f2bf_pad_kernel<<<(nWoPad + 255) / 256, 256, 0, stream>>>(Wo, wo_bf, nWo, nWoPad);

    // ---- 2. embeddings
    embed_pos_kernel<<<kB * kS, kH, 0, stream>>>(contexts, emb, ctx_bf);
    embed_q_kernel<<<kB * kTQ, kH, 0, stream>>>(questions, emb, qemb_bf);

    // ---- 3. input-gate GEMMs (all timesteps at once; no recurrence in gi)
    dim3 gCtx(kH3 / 256, (kB * kS) / 32);    // (3, 200)
    wmma_gemm_bias_kernel<<<gCtx, 128, 0, stream>>>(ctx_bf, wihf_bf, bih_f, gi_f,
                                                    kB * kS, kH3, kH);
    wmma_gemm_bias_kernel<<<gCtx, 128, 0, stream>>>(ctx_bf, wihb_bf, bih_b, gi_b,
                                                    kB * kS, kH3, kH);
    dim3 gQ(kH3 / 256, (kB * kTQ) / 32);     // (3, 128)
    wmma_gemm_bias_kernel<<<gQ, 128, 0, stream>>>(qemb_bf, wihq_bf, bih_q, gi_q,
                                                  kB * kTQ, kH3, kH);

    // ---- 4. zero hidden states
    const int nH3 = 3 * kB * kH;
    init_h_kernel<<<(nH3 + 255) / 256, 256, 0, stream>>>(hf_all, hb_all, nH3);

    // ---- 5. bidirectional context GRUs (recurrent part, one launch per step)
    for (int t = 0; t < kS; ++t) {
        gru_step_kernel<<<kB / 16, 512, 0, stream>>>(
            hf_f, hb_f, whhf_bf, bhh_f,
            gi_f + (size_t)t * kH3, (long)(kS * kH3),
            out_f + (size_t)t * kH, (long)(kS * kH));
        const int s = kS - 1 - t;            // backward GRU: read & write reversed slot
        gru_step_kernel<<<kB / 16, 512, 0, stream>>>(
            hf_b, hb_b, whhb_bf, bhh_b,
            gi_b + (size_t)s * kH3, (long)(kS * kH3),
            out_b + (size_t)s * kH, (long)(kS * kH));
    }

    // ---- 6. question GRU (keep only last hidden)
    for (int t = 0; t < kTQ; ++t) {
        gru_step_kernel<<<kB / 16, 512, 0, stream>>>(
            hf_q, hb_q, whhq_bf, bhh_q,
            gi_q + (size_t)t * kH3, (long)(kTQ * kH3),
            nullptr, 0L);
    }

    // ---- 7. facts = out_f + out_b(reversed-in-place)
    const int nF = kB * kS * kH;
    add_kernel<<<(nF + 255) / 256, 256, 0, stream>>>(out_f, out_b, facts, nF);

    // ---- 8. attention + read + z = tanh(q*read)
    attn_kernel<<<kB, kH, 0, stream>>>(facts, hf_q, z_bf);

    // ---- 9. vocab projection: out = z @ Wo^T + bo  (128 x 50257, K=256)
    dim3 gOut(kVpad / 256, kB / 32);         // (197, 4)
    wmma_gemm_bias_kernel<<<gOut, 128, 0, stream>>>(z_bf, wo_bf, bo, (float*)d_out,
                                                    kB, kV, kH);
}
